// MultiAttention_5480378270189
// MI455X (gfx1250) — matrix operand order repaired and checked by reading, not compiled, not run on hardware
//
#include <hip/hip_runtime.h>
#include <hip/hip_bf16.h>

// ---------------------------------------------------------------------------
// MultiAttention on gfx1250 (MI455X): bf16 WMMA pipeline with CDNA5
// async-to-LDS double-buffered staging.
//   k0a: f32 -> bf16 (activations)
//   k0b: f32 -> bf16 transposed (weights: W[K,N] -> WT[N,K])
//   k0c: mask int32 [S,S] -> packed bits [S, S/32]
//   k1 : tiled GEMM+bias (bf16, f32 acc), optional transposed store (for V^T)
//   k2 : flash attention (online softmax, packed-bit mask, async K/V^T tiles)
//   k3 : k1 with f32 output -> final projection
// ---------------------------------------------------------------------------

typedef __attribute__((ext_vector_type(16))) __bf16 v16bf;
typedef __attribute__((ext_vector_type(8)))  __bf16 v8bf;
typedef __attribute__((ext_vector_type(8)))  float  v8f;

static constexpr int Bn = 4, Sn = 1024, Dn = 1024, Hn = 16, DKn = 64;

static __device__ inline v8f wmma_bf16(v16bf a, v16bf b, v8f c) {
    return __builtin_amdgcn_wmma_f32_16x16x32_bf16(false, a, false, b, (short)0, c,
                                                   false, false);
}

static __device__ inline v16bf ld16(const __bf16* p0, const __bf16* p1) {
    v8bf lo = *(const v8bf*)p0;
    v8bf hi = *(const v8bf*)p1;
    v16bf r;
#pragma unroll
    for (int i = 0; i < 8; ++i) { r[i] = lo[i]; r[i + 8] = hi[i]; }
    return r;
}

static __device__ inline v8f vzero8() {
    v8f v;
#pragma unroll
    for (int i = 0; i < 8; ++i) v[i] = 0.0f;
    return v;
}

// Generic LDS pointer -> 32-bit LDS byte offset (low 32 bits of the shared
// aperture address are the LDS offset per the CDNA5 aperture rules).
static __device__ inline unsigned lds_off(const void* p) {
    return (unsigned)(uintptr_t)p;
}

// CDNA5 async global->LDS copies (ASYNCcnt tracked). Per-lane: dst LDS byte
// offset in a VGPR, 64-bit global address in a VGPR pair.
static __device__ inline void async_b128(unsigned lds, const void* gaddr) {
    asm volatile("global_load_async_to_lds_b128 %0, %1, off"
                 :: "v"(lds), "v"(gaddr) : "memory");
}
static __device__ inline void async_b32(unsigned lds, const void* gaddr) {
    asm volatile("global_load_async_to_lds_b32 %0, %1, off"
                 :: "v"(lds), "v"(gaddr) : "memory");
}
static __device__ inline void wait_async0() {
    asm volatile("s_wait_asynccnt 0" ::: "memory");
}

// ---------------------------------------------------------------------------
// k0a: elementwise f32 -> bf16
// ---------------------------------------------------------------------------
__global__ void cvt_f32_bf16(const float* __restrict__ s, __bf16* __restrict__ d, int n) {
    int i = blockIdx.x * blockDim.x + threadIdx.x;
    int stride = gridDim.x * blockDim.x;
    for (; i < n; i += stride) d[i] = (__bf16)s[i];
}

// ---------------------------------------------------------------------------
// k0b: batched transpose + convert: src[b][k][n] f32 -> dst[b][n][k] bf16
// ---------------------------------------------------------------------------
__global__ void cvtT_f32_bf16(const float* __restrict__ s, __bf16* __restrict__ d,
                              int K, int N, int total) {
    int i = blockIdx.x * blockDim.x + threadIdx.x;
    int stride = gridDim.x * blockDim.x;
    const int kn = K * N;
    for (; i < total; i += stride) {
        int b = i / kn;
        int r = i - b * kn;
        int n = r / K;
        int k = r - n * K;
        d[i] = (__bf16)s[(size_t)b * kn + (size_t)k * N + n];
    }
}

// ---------------------------------------------------------------------------
// k0c: pack mask [S,S] int32 -> bits [S, S/32] u32 (bit j of word w = mask col)
// ---------------------------------------------------------------------------
__global__ void pack_mask(const int* __restrict__ m, unsigned* __restrict__ bits) {
    int i = blockIdx.x * blockDim.x + threadIdx.x;          // word index
    const int nwords = Sn * (Sn / 32);
    int stride = gridDim.x * blockDim.x;
    for (; i < nwords; i += stride) {
        const int q = i >> 5;
        const int w = i & 31;
        unsigned acc = 0u;
#pragma unroll
        for (int j = 0; j < 32; ++j)
            acc |= (m[(size_t)q * Sn + w * 32 + j] != 0 ? 1u : 0u) << j;
        bits[i] = acc;
    }
}

// ---------------------------------------------------------------------------
// k1/k3: C = A[M,K] @ Bt[N,K]^T + bias. A bf16 row-major, Bt = pre-transposed
// B (row n holds column n of B, contiguous in k). Block = 256 threads,
// tile 128x64, K step 32, async double-buffered Bt tile in LDS.
// TSTORE=true writes C transposed (C[n][m], ldc = row stride in n).
// ---------------------------------------------------------------------------
template <typename OutT, bool TSTORE>
__global__ __launch_bounds__(256) void gemm_bias_kernel(
    const __bf16* __restrict__ A, size_t sAz,
    const __bf16* __restrict__ Bt, size_t sBy, int lda, int ldbt,
    const float* __restrict__ bias, int sBiasY,
    OutT* __restrict__ C, size_t sCz, size_t sCy, int ldc, int K) {

    const __bf16* A_blk = A + (size_t)blockIdx.z * sAz;
    const __bf16* B_blk = Bt + (size_t)blockIdx.y * sBy;
    const float*  bias_blk = bias + (size_t)blockIdx.y * sBiasY;
    OutT* C_blk = C + (size_t)blockIdx.z * sCz + (size_t)blockIdx.y * sCy;

    const int tid  = threadIdx.x;
    const int w    = tid >> 5;
    const int lane = tid & 31;
    const int nl   = lane & 15;
    const int hi   = lane >> 4;
    const int row0 = blockIdx.x * 128;

    __shared__ __bf16 BT[2][64 * 48];   // Bt tile [n][k], stride 48

    // per-thread async copy slot: row n = tid>>2 (0..63), chunk c = tid&3
    const int bn = tid >> 2, bc = tid & 3;
    const __bf16* bsrc0 = B_blk + (size_t)bn * ldbt + bc * 8;
    const unsigned bdst[2] = { lds_off(&BT[0][bn * 48 + bc * 8]),
                               lds_off(&BT[1][bn * 48 + bc * 8]) };

    auto issue = [&](int kc, int buf) {
        async_b128(bdst[buf], bsrc0 + kc);
    };

    v8f acc[4];
#pragma unroll
    for (int t = 0; t < 4; ++t) acc[t] = vzero8();

    issue(0, 0);

    const __bf16* arow = A_blk + (size_t)(row0 + w * 16 + nl) * lda + hi * 8;

    const int nsteps = K / 32;
    for (int i = 0; i < nsteps; ++i) {
        const int kc = i * 32;
        const int buf = i & 1;
        wait_async0();
        __syncthreads();
        if (i + 1 < nsteps) issue(kc + 32, buf ^ 1);

        const __bf16* ar = arow + kc;
        if (i + 1 < nsteps) __builtin_prefetch(ar + 32, 0, 0);
        v16bf a = ld16(ar, ar + 16);

#pragma unroll
        for (int t = 0; t < 4; ++t) {
            const __bf16* bp0 = &BT[buf][(16 * t + nl) * 48 + hi * 8];
            v16bf bf = ld16(bp0, bp0 + 16);
            acc[t] = wmma_bf16(a, bf, acc[t]);
        }
        __syncthreads();
    }

#pragma unroll
    for (int t = 0; t < 4; ++t) {
        const int col = 16 * t + nl;
        const float bv = bias_blk[col];
#pragma unroll
        for (int v = 0; v < 8; ++v) {
            const int m = row0 + w * 16 + v + 8 * hi;
            const float r = acc[t][v] + bv;
            if (TSTORE) C_blk[(size_t)col * ldc + m] = (OutT)r;
            else        C_blk[(size_t)m * ldc + col] = (OutT)r;
        }
    }
}

// ---------------------------------------------------------------------------
// k2: flash attention. grid = (S/128, H, B), block = 256 (8 waves).
// Q,K: bf16 [B,H,S,64]; Vt: bf16 [B,H,64,S]; mask packed bits [S,S/32].
// K tile / V^T tile / mask words double-buffered via async-to-LDS.
// ---------------------------------------------------------------------------
__global__ __launch_bounds__(256) void flash_attn_kernel(
    const __bf16* __restrict__ Qp, const __bf16* __restrict__ Kp,
    const __bf16* __restrict__ Vt, const unsigned* __restrict__ mbits,
    __bf16* __restrict__ Ocat) {

    const int b = blockIdx.z, h = blockIdx.y, qt = blockIdx.x;
    const size_t bh = ((size_t)b * Hn + h) * Sn * DKn;
    const __bf16* Qb = Qp + bh;
    const __bf16* Kb = Kp + bh;
    const __bf16* Vb = Vt + bh;          // [64][S]

    const int tid  = threadIdx.x;
    const int w    = tid >> 5;
    const int lane = tid & 31;
    const int nl   = lane & 15;
    const int hi   = lane >> 4;
    const int q0b  = qt * 128;
    const int q0   = q0b + w * 16;

    __shared__ __bf16 Kl[2][32 * 72];    // K tile [k][d], stride 72
    __shared__ __bf16 VTl[2][64 * 48];   // V^T tile [v][k], stride 48
    __shared__ unsigned Ml[2][128];      // mask words, one per q row
    __shared__ __bf16 Pl[8][16 * 48];    // per-wave P tile [q][k], stride 48

    // async copy slots
    const int kk = tid >> 3, kn0 = (tid & 7) * 8;        // K tile: 32x64
    const __bf16* ksrc0 = Kb + (size_t)kk * DKn + kn0;
    const unsigned kdst[2] = { lds_off(&Kl[0][kk * 72 + kn0]),
                               lds_off(&Kl[1][kk * 72 + kn0]) };
    const int vr = tid >> 2, vc = tid & 3;               // V^T tile: 64x32
    const __bf16* vsrc0 = Vb + (size_t)vr * Sn + vc * 8;
    const unsigned vdst[2] = { lds_off(&VTl[0][vr * 48 + vc * 8]),
                               lds_off(&VTl[1][vr * 48 + vc * 8]) };
    const unsigned* msrc0 = mbits + (size_t)(q0b + (tid & 127)) * (Sn / 32);
    const unsigned mdst[2] = { lds_off(&Ml[0][tid & 127]),
                               lds_off(&Ml[1][tid & 127]) };

    auto issue = [&](int kt, int buf) {
        async_b128(kdst[buf], ksrc0 + kt * 32 * DKn);
        async_b128(vdst[buf], vsrc0 + kt * 32);
        if (tid < 128) async_b32(mdst[buf], msrc0 + kt);
    };

    // Q fragments for this wave's 16 rows
    const __bf16* qr = Qb + (size_t)(q0 + nl) * DKn + hi * 8;
    const v16bf qf0 = ld16(qr, qr + 16);
    const v16bf qf1 = ld16(qr + 32, qr + 48);

    v8f acc[4];
#pragma unroll
    for (int t = 0; t < 4; ++t) acc[t] = vzero8();
    float mrow[8], lrow[8];
#pragma unroll
    for (int v = 0; v < 8; ++v) { mrow[v] = -INFINITY; lrow[v] = 0.0f; }

    issue(0, 0);

    const int NT = Sn / 32;
    for (int kt = 0; kt < NT; ++kt) {
        const int buf = kt & 1;
        wait_async0();
        __syncthreads();
        if (kt + 1 < NT) issue(kt + 1, buf ^ 1);

        // scores: 16q x 32k, accumulate over d = 64
        v8f s[2];
#pragma unroll
        for (int g = 0; g < 2; ++g) {
            s[g] = vzero8();
            const __bf16* kp0 = &Kl[buf][(16 * g + nl) * 72 + hi * 8];
            v16bf bk0 = ld16(kp0, kp0 + 16);
            v16bf bk1 = ld16(kp0 + 32, kp0 + 48);
            s[g] = wmma_bf16(qf0, bk0, s[g]);
            s[g] = wmma_bf16(qf1, bk1, s[g]);
        }
        // mask from packed bits (one word per q row covers this 32-key tile)
#pragma unroll
        for (int v = 0; v < 8; ++v) {
            const unsigned word = Ml[buf][w * 16 + v + 8 * hi];
            s[0][v] = ((word >> nl) & 1u)        ? s[0][v] : -1.0e9f;
            s[1][v] = ((word >> (16 + nl)) & 1u) ? s[1][v] : -1.0e9f;
        }

        // online softmax (row stats replicated across 16-lane halves)
#pragma unroll
        for (int v = 0; v < 8; ++v) {
            float x = fmaxf(s[0][v], s[1][v]);
#pragma unroll
            for (int d = 1; d < 16; d <<= 1) x = fmaxf(x, __shfl_xor(x, d, 32));
            const float mn = fmaxf(mrow[v], x);
            const float al = __expf(mrow[v] - mn);
            mrow[v] = mn;
            const float p0 = __expf(s[0][v] - mn);
            const float p1 = __expf(s[1][v] - mn);
            s[0][v] = p0; s[1][v] = p1;
            float rs = p0 + p1;
#pragma unroll
            for (int d = 1; d < 16; d <<= 1) rs += __shfl_xor(rs, d, 32);
            lrow[v] = lrow[v] * al + rs;
#pragma unroll
            for (int t = 0; t < 4; ++t) acc[t][v] *= al;
        }

        // P: C-layout -> per-wave LDS slab -> A-layout (same-wave DS ordering)
        __bf16* Pw = Pl[w];
#pragma unroll
        for (int g = 0; g < 2; ++g)
#pragma unroll
            for (int v = 0; v < 8; ++v)
                Pw[(v + 8 * hi) * 48 + 16 * g + nl] = (__bf16)s[g][v];

        const __bf16* pr = Pw + nl * 48 + hi * 8;
        v16bf pf = ld16(pr, pr + 16);

#pragma unroll
        for (int t = 0; t < 4; ++t) {
            const __bf16* vp0 = &VTl[buf][(16 * t + nl) * 48 + hi * 8];
            v16bf bv = ld16(vp0, vp0 + 16);
            acc[t] = wmma_bf16(pf, bv, acc[t]);
        }
        __syncthreads();
    }

    // normalize and write concatenated heads: Ocat[b, q, h*64 + col]
#pragma unroll
    for (int v = 0; v < 8; ++v) {
        const float inv = 1.0f / lrow[v];
        const int q = q0 + v + 8 * hi;
        const size_t base = ((size_t)b * Sn + q) * (Hn * DKn) + h * DKn;
#pragma unroll
        for (int t = 0; t < 4; ++t)
            Ocat[base + 16 * t + nl] = (__bf16)(acc[t][v] * inv);
    }
}

// ---------------------------------------------------------------------------
// launch
// ---------------------------------------------------------------------------
extern "C" void kernel_launch(void* const* d_in, const int* in_sizes, int n_in,
                              void* d_out, int out_size, void* d_ws, size_t ws_size,
                              hipStream_t stream) {
    const float* querys = (const float*)d_in[0];
    const float* keys   = (const float*)d_in[1];
    const float* values = (const float*)d_in[2];
    const int*   mask   = (const int*)d_in[3];
    const float* Wq = (const float*)d_in[4];
    const float* bq = (const float*)d_in[5];
    const float* Wk = (const float*)d_in[6];
    const float* bk = (const float*)d_in[7];
    const float* Wv = (const float*)d_in[8];
    const float* bv = (const float*)d_in[9];
    const float* Wp = (const float*)d_in[10];
    const float* bp = (const float*)d_in[11];
    float* out = (float*)d_out;

    const size_t NX = (size_t)Bn * Sn * Dn;        // 4M
    const size_t NW = (size_t)Hn * Dn * DKn;       // 1M
    const size_t NWP = (size_t)(Hn * DKn) * Dn;    // 1M
    const size_t NP = (size_t)Bn * Hn * Sn * DKn;  // 4M

    __bf16* wb   = (__bf16*)d_ws;
    __bf16* Xq   = wb;
    __bf16* Xk   = Xq + NX;
    __bf16* Xv   = Xk + NX;
    __bf16* WqT  = Xv + NX;       // [H][64][D]
    __bf16* WkT  = WqT + NW;
    __bf16* WvT  = WkT + NW;
    __bf16* WpT  = WvT + NW;      // [1024][1024] (N x K)
    __bf16* Qp   = WpT + NWP;     // [B,H,S,64]
    __bf16* Kp   = Qp + NP;
    __bf16* Vtg  = Kp + NP;       // [B,H,64,S]
    __bf16* Oc   = Vtg + NP;      // [B,S,1024]
    unsigned* Mb = (unsigned*)(Oc + NX);  // [S, S/32]

    const dim3 blk(256);
    cvt_f32_bf16<<<2048, blk, 0, stream>>>(querys, Xq, (int)NX);
    cvt_f32_bf16<<<2048, blk, 0, stream>>>(keys,   Xk, (int)NX);
    cvt_f32_bf16<<<2048, blk, 0, stream>>>(values, Xv, (int)NX);
    cvtT_f32_bf16<<<1024, blk, 0, stream>>>(Wq, WqT, Dn, DKn, (int)NW);
    cvtT_f32_bf16<<<1024, blk, 0, stream>>>(Wk, WkT, Dn, DKn, (int)NW);
    cvtT_f32_bf16<<<1024, blk, 0, stream>>>(Wv, WvT, Dn, DKn, (int)NW);
    cvtT_f32_bf16<<<1024, blk, 0, stream>>>(Wp, WpT, Hn * DKn, Dn, (int)NWP);
    pack_mask<<<128, blk, 0, stream>>>(mask, Mb);

    // QKV projections: grid (S/128, H, B)
    const dim3 gp(Sn / 128, Hn, Bn);
    gemm_bias_kernel<__bf16, false><<<gp, blk, 0, stream>>>(
        Xq, (size_t)Sn * Dn, WqT, (size_t)DKn * Dn, Dn, Dn,
        bq, DKn, Qp, (size_t)Hn * Sn * DKn, (size_t)Sn * DKn, DKn, Dn);
    gemm_bias_kernel<__bf16, false><<<gp, blk, 0, stream>>>(
        Xk, (size_t)Sn * Dn, WkT, (size_t)DKn * Dn, Dn, Dn,
        bk, DKn, Kp, (size_t)Hn * Sn * DKn, (size_t)Sn * DKn, DKn, Dn);
    // V projection stores transposed: Vtg[b,h,v,s]
    gemm_bias_kernel<__bf16, true><<<gp, blk, 0, stream>>>(
        Xv, (size_t)Sn * Dn, WvT, (size_t)DKn * Dn, Dn, Dn,
        bv, DKn, Vtg, (size_t)Hn * DKn * Sn, (size_t)DKn * Sn, Sn, Dn);

    // flash attention
    flash_attn_kernel<<<gp, blk, 0, stream>>>(Qp, Kp, Vtg, Mb, Oc);

    // output projection: [B*S, 1024] @ Wp + bp -> f32
    const dim3 go((Bn * Sn) / 128, (Hn * DKn) / 64, 1);
    gemm_bias_kernel<float, false><<<go, blk, 0, stream>>>(
        Oc, 0, WpT, (size_t)64 * Dn, Hn * DKn, Hn * DKn,
        bp, 64, out, 0, (size_t)64, Dn, Hn * DKn);
}
